// STA_41712722379245
// MI455X (gfx1250) — compile-verified, matrix operand order checked by reading, not run on hardware
//
#include <hip/hip_runtime.h>
#include <hip/hip_bf16.h>

// ---------------------------------------------------------------------------
// Problem constants (from reference): B=32, C=1024, IC=512, T=8, H*W=128
// ---------------------------------------------------------------------------
#define NB   32
#define NC   1024
#define NIC  512
#define NT   8
#define NHW  128
#define NBT  (NB * NT)        // 256
#define NP   (NBT * NHW)      // 32768 pixels

typedef __bf16 bf16_t;
typedef __attribute__((ext_vector_type(16))) __bf16 v16bf;
typedef __attribute__((ext_vector_type(8)))  __bf16 v8bf;
typedef __attribute__((ext_vector_type(4)))  __bf16 v4bf;
typedef __attribute__((ext_vector_type(8)))  float  v8f;

__device__ __forceinline__ bf16_t f2bf(float f) { return (bf16_t)f; }

// Load one 16x32 bf16 WMMA fragment (A, or B given as B^T row-major [n][k]).
// Per CDNA5 ISA 16-bit A layout: lanes 0-15 -> row m, K=0..7 (VGPR0-3) and
// K=16..23 (VGPR4-7); lanes 16-31 -> same rows, K=8..15 and K=24..31.
// Rows are K-contiguous in memory, so each lane does two 16-byte loads.
__device__ __forceinline__ v16bf load_frag_bf16(const bf16_t* __restrict__ base,
                                                int ld, int lane) {
  const int r  = lane & 15;
  const int k0 = (lane >> 4) << 3;            // 0 or 8
  const bf16_t* p = base + (size_t)r * ld + k0;
  v8bf lo = *reinterpret_cast<const v8bf*>(p);        // K = k0 .. k0+7
  v8bf hi = *reinterpret_cast<const v8bf*>(p + 16);   // K = k0+16 .. k0+23
  v16bf out;
#pragma unroll
  for (int i = 0; i < 8; ++i) { out[i] = lo[i]; out[i + 8] = hi[i]; }
  return out;
}

__device__ __forceinline__ v8f wmma_bf16(v16bf a, v16bf b, v8f c) {
  return __builtin_amdgcn_wmma_f32_16x16x32_bf16(false, a, false, b,
                                                 (short)0, c, false, false);
}

// ---------------------------------------------------------------------------
// K0: fp32 -> bf16 copy (weights)
// ---------------------------------------------------------------------------
__global__ void k_cvt_bf16(const float* __restrict__ in, bf16_t* __restrict__ out, int n) {
  int i = blockIdx.x * blockDim.x + threadIdx.x;
  if (i < n) out[i] = f2bf(in[i]);
}

// ---------------------------------------------------------------------------
// K1: transpose x (b,c,t,hw) -> xbf[p][c] bf16 (p = (b*T+t)*128+hw), fused
//     with spatial mean pooling -> xpbf[(b*T+t)][c] bf16.
// grid: (C/32, B*T), block: 256
// ---------------------------------------------------------------------------
__global__ void k_transpose_pool(const float* __restrict__ x,
                                 bf16_t* __restrict__ xbf,
                                 bf16_t* __restrict__ xpbf) {
  __shared__ float tile[32][NHW + 1];
  __shared__ float psum[32][8];
  const int cb  = blockIdx.x;       // channel block (32 channels)
  const int bt  = blockIdx.y;       // 0..255
  const int b   = bt >> 3;
  const int t   = bt & 7;
  const int tid = threadIdx.x;

  // coalesced load: 8 threads per channel row, 16 floats each
  const int ci  = tid >> 3;
  const int hw0 = (tid & 7) * 16;
  const float* src = x + (((size_t)(b * NC + cb * 32 + ci) * NT + t) * NHW) + hw0;
  float s = 0.f;
#pragma unroll
  for (int j = 0; j < 16; j += 4) {
    float4 v = *reinterpret_cast<const float4*>(src + j);
    tile[ci][hw0 + j + 0] = v.x;
    tile[ci][hw0 + j + 1] = v.y;
    tile[ci][hw0 + j + 2] = v.z;
    tile[ci][hw0 + j + 3] = v.w;
    s += v.x + v.y + v.z + v.w;
  }
  psum[ci][tid & 7] = s;
  __syncthreads();

  // pooled mean -> bf16
  if (tid < 32) {
    float tot = 0.f;
#pragma unroll
    for (int j = 0; j < 8; ++j) tot += psum[tid][j];
    xpbf[(size_t)bt * NC + cb * 32 + tid] = f2bf(tot * (1.0f / 128.0f));
  }

  // transposed write: each thread writes 16 contiguous channels of one pixel
  const int hw  = tid >> 1;
  const int ch0 = (tid & 1) * 16;
  bf16_t* dst = xbf + ((size_t)(bt * NHW + hw)) * NC + cb * 32 + ch0;
  v8bf o0, o1;
#pragma unroll
  for (int j = 0; j < 8; ++j) {
    o0[j] = f2bf(tile[ch0 + j][hw]);
    o1[j] = f2bf(tile[ch0 + 8 + j][hw]);
  }
  *reinterpret_cast<v8bf*>(dst)     = o0;
  *reinterpret_cast<v8bf*>(dst + 8) = o1;
}

// ---------------------------------------------------------------------------
// K2: theta / phi GEMM. M=256 (b*t), N=512 (ic), K=1024 (c), bf16 WMMA.
// grid: (N/16, M/16, 2) ; block: 32 (one wave per 16x16 tile)
// ---------------------------------------------------------------------------
__global__ void k_qk_gemm(const bf16_t* __restrict__ xpbf,
                          const bf16_t* __restrict__ twbf,
                          const bf16_t* __restrict__ pwbf,
                          const float* __restrict__ tb,
                          const float* __restrict__ pb,
                          float* __restrict__ theta,
                          float* __restrict__ phi) {
  const int nt   = blockIdx.x;
  const int mt   = blockIdx.y;
  const int sel  = blockIdx.z;
  const bf16_t* W  = sel ? pwbf : twbf;
  const float*  bs = sel ? pb   : tb;
  float*        O  = sel ? phi  : theta;
  const int lane = threadIdx.x & 31;

  const bf16_t* A  = xpbf + (size_t)mt * 16 * NC;
  const bf16_t* Bm = W    + (size_t)nt * 16 * NC;
  v8f acc = {};
#pragma unroll 4
  for (int k = 0; k < NC; k += 32) {
    v16bf a = load_frag_bf16(A + k,  NC, lane);
    v16bf b = load_frag_bf16(Bm + k, NC, lane);
    acc = wmma_bf16(a, b, acc);
  }
  const int n  = nt * 16 + (lane & 15);
  const int m0 = mt * 16 + ((lane >> 4) << 3);
  const float bv = bs[n];
#pragma unroll
  for (int v = 0; v < 8; ++v)
    O[(size_t)(m0 + v) * NIC + n] = acc[v] + bv;
}

// ---------------------------------------------------------------------------
// K3: f = theta . phi (per batch, 8x8, K=512), softmax rows -> attn
// grid: B ; block: 64 (thread = (t,s))
// ---------------------------------------------------------------------------
__global__ void k_attn(const float* __restrict__ theta,
                       const float* __restrict__ phi,
                       float* __restrict__ attn) {
  const int b   = blockIdx.x;
  const int tid = threadIdx.x;
  const int t   = tid >> 3;
  const int s   = tid & 7;
  const float* th = theta + (size_t)(b * NT + t) * NIC;
  const float* ph = phi   + (size_t)(b * NT + s) * NIC;
  float f = 0.f;
  for (int ic = 0; ic < NIC; ++ic) f += th[ic] * ph[ic];

  __shared__ float fs[64];
  __shared__ float es[64];
  fs[tid] = f;
  __syncthreads();
  float mx = -1e30f;
#pragma unroll
  for (int j = 0; j < 8; ++j) mx = fmaxf(mx, fs[t * 8 + j]);
  float e = expf(f - mx);
  es[tid] = e;
  __syncthreads();
  float sum = 0.f;
#pragma unroll
  for (int j = 0; j < 8; ++j) sum += es[t * 8 + j];
  attn[(size_t)b * 64 + tid] = e / sum;
}

// ---------------------------------------------------------------------------
// K4: g projection GEMM, register-blocked.
//     D[m=ic][n=pixel]:  A = g_w bf16 (rows ic, K=c), B = xbf (rows p, K=c).
//     Wave computes 2(m) x 4(n) tiles = 32 ic x 64 pixels, 8 WMMAs / K-chunk.
//     Per-lane acc runs along ic (contiguous in g[p][ic]) -> 16B v8bf stores.
// grid: (NP/64, NIC/32/8) ; block: 256 (8 waves along ic)
// ---------------------------------------------------------------------------
__global__ void k_g_gemm(const bf16_t* __restrict__ xbf,
                         const bf16_t* __restrict__ gwbf,
                         const float* __restrict__ gb,
                         bf16_t* __restrict__ g) {
  const int wave  = threadIdx.x >> 5;
  const int lane  = threadIdx.x & 31;
  const int pbase = blockIdx.x * 64;                       // pixel base (N)
  const int icb   = (blockIdx.y * 8 + wave) * 32;          // ic base (M)

  v8f acc[2][4] = {};
  for (int k = 0; k < NC; k += 32) {
    v16bf bfr[4];
#pragma unroll
    for (int ni = 0; ni < 4; ++ni)
      bfr[ni] = load_frag_bf16(xbf + (size_t)(pbase + ni * 16) * NC + k, NC, lane);
#pragma unroll
    for (int mi = 0; mi < 2; ++mi) {
      v16bf a = load_frag_bf16(gwbf + (size_t)(icb + mi * 16) * NC + k, NC, lane);
#pragma unroll
      for (int ni = 0; ni < 4; ++ni)
        acc[mi][ni] = wmma_bf16(a, bfr[ni], acc[mi][ni]);
    }
  }

  // epilogue: per lane, 8 consecutive ic values of one pixel column
  const int n  = lane & 15;
  const int m0 = (lane >> 4) << 3;
#pragma unroll
  for (int mi = 0; mi < 2; ++mi) {
    const int ic0 = icb + mi * 16 + m0;
    float4 b0 = *reinterpret_cast<const float4*>(gb + ic0);
    float4 b1 = *reinterpret_cast<const float4*>(gb + ic0 + 4);
#pragma unroll
    for (int ni = 0; ni < 4; ++ni) {
      const int p = pbase + ni * 16 + n;
      v8bf o;
      o[0] = f2bf(acc[mi][ni][0] + b0.x);
      o[1] = f2bf(acc[mi][ni][1] + b0.y);
      o[2] = f2bf(acc[mi][ni][2] + b0.z);
      o[3] = f2bf(acc[mi][ni][3] + b0.w);
      o[4] = f2bf(acc[mi][ni][4] + b1.x);
      o[5] = f2bf(acc[mi][ni][5] + b1.y);
      o[6] = f2bf(acc[mi][ni][6] + b1.z);
      o[7] = f2bf(acc[mi][ni][7] + b1.w);
      *reinterpret_cast<v8bf*>(g + (size_t)p * NIC + ic0) = o;
    }
  }
}

// ---------------------------------------------------------------------------
// K5: temporal mix  y[b,t,hw,ic] = sum_s attn[b,t,s] * g[b,s,hw,ic]
// grid: (128*128/256, B) ; block 256 ; 4 ic per thread
// ---------------------------------------------------------------------------
__global__ void k_mix(const bf16_t* __restrict__ g,
                      const float* __restrict__ attn,
                      bf16_t* __restrict__ y) {
  const int b = blockIdx.y;
  __shared__ float a_s[64];
  if (threadIdx.x < 64) a_s[threadIdx.x] = attn[(size_t)b * 64 + threadIdx.x];
  __syncthreads();

  const int idx = blockIdx.x * blockDim.x + threadIdx.x;  // 0..16383
  const int ic0 = (idx & 127) * 4;
  const int hw  = idx >> 7;

  float acc[NT][4] = {};
#pragma unroll
  for (int s = 0; s < NT; ++s) {
    const bf16_t* gp = g + (size_t)((b * NT + s) * NHW + hw) * NIC + ic0;
    v4bf gv = *reinterpret_cast<const v4bf*>(gp);
    float g0 = (float)gv[0], g1 = (float)gv[1], g2 = (float)gv[2], g3 = (float)gv[3];
#pragma unroll
    for (int t = 0; t < NT; ++t) {
      const float av = a_s[t * 8 + s];
      acc[t][0] += av * g0; acc[t][1] += av * g1;
      acc[t][2] += av * g2; acc[t][3] += av * g3;
    }
  }
#pragma unroll
  for (int t = 0; t < NT; ++t) {
    v4bf o;
#pragma unroll
    for (int j = 0; j < 4; ++j) o[j] = f2bf(acc[t][j]);
    *reinterpret_cast<v4bf*>(y + (size_t)((b * NT + t) * NHW + hw) * NIC + ic0) = o;
  }
}

// ---------------------------------------------------------------------------
// K6: output projection GEMM + bias + BN affine + residual, register-blocked.
//     A = ybuf (rows p, K=ic), B = w_w bf16 (rows c, K=ic).
//     Wave computes 4(m) x 2(n) tiles = 64 pixels x 32 c, 8 WMMAs / K-chunk.
//     Per-lane acc runs along hw (contiguous in output) -> float4 I/O.
// grid: (NC/32, NP/64/8) ; block: 256 (8 waves along pixels)
// ---------------------------------------------------------------------------
__global__ void k_out_gemm(const bf16_t* __restrict__ y,
                           const bf16_t* __restrict__ wwbf,
                           const float* __restrict__ wb,
                           const float* __restrict__ bn_g,
                           const float* __restrict__ bn_b,
                           const float* __restrict__ bn_m,
                           const float* __restrict__ bn_v,
                           const float* __restrict__ x,
                           float* __restrict__ out) {
  const int wave  = threadIdx.x >> 5;
  const int lane  = threadIdx.x & 31;
  const int cb    = blockIdx.x * 32;                       // c base (N)
  const int pbase = (blockIdx.y * 8 + wave) * 64;          // pixel base (M)

  v8f acc[4][2] = {};
  for (int k = 0; k < NIC; k += 32) {
    v16bf bfr[2];
#pragma unroll
    for (int ni = 0; ni < 2; ++ni)
      bfr[ni] = load_frag_bf16(wwbf + (size_t)(cb + ni * 16) * NIC + k, NIC, lane);
#pragma unroll
    for (int mi = 0; mi < 4; ++mi) {
      v16bf a = load_frag_bf16(y + (size_t)(pbase + mi * 16) * NIC + k, NIC, lane);
#pragma unroll
      for (int ni = 0; ni < 2; ++ni)
        acc[mi][ni] = wmma_bf16(a, bfr[ni], acc[mi][ni]);
    }
  }

  const int n  = lane & 15;
  const int m0 = (lane >> 4) << 3;
#pragma unroll
  for (int ni = 0; ni < 2; ++ni) {
    const int c = cb + ni * 16 + n;
    const float inv   = bn_g[c] * rsqrtf(bn_v[c] + 1e-5f);
    const float shift = bn_b[c] - bn_m[c] * inv;
    const float bias  = wb[c];
#pragma unroll
    for (int mi = 0; mi < 4; ++mi) {
      const int p  = pbase + mi * 16 + m0;   // 8 consecutive pixels (same b,t)
      const int b  = p >> 10;
      const int t  = (p >> 7) & 7;
      const int hw = p & 127;
      const size_t base = (((size_t)(b * NC + c) * NT + t) * NHW) + hw;
      float4 x0 = *reinterpret_cast<const float4*>(x + base);
      float4 x1 = *reinterpret_cast<const float4*>(x + base + 4);
      float4 r0, r1;
      r0.x = (acc[mi][ni][0] + bias) * inv + shift + x0.x;
      r0.y = (acc[mi][ni][1] + bias) * inv + shift + x0.y;
      r0.z = (acc[mi][ni][2] + bias) * inv + shift + x0.z;
      r0.w = (acc[mi][ni][3] + bias) * inv + shift + x0.w;
      r1.x = (acc[mi][ni][4] + bias) * inv + shift + x1.x;
      r1.y = (acc[mi][ni][5] + bias) * inv + shift + x1.y;
      r1.z = (acc[mi][ni][6] + bias) * inv + shift + x1.z;
      r1.w = (acc[mi][ni][7] + bias) * inv + shift + x1.w;
      *reinterpret_cast<float4*>(out + base)     = r0;
      *reinterpret_cast<float4*>(out + base + 4) = r1;
    }
  }
}

// ---------------------------------------------------------------------------
// Host launcher
// ---------------------------------------------------------------------------
extern "C" void kernel_launch(void* const* d_in, const int* in_sizes, int n_in,
                              void* d_out, int out_size, void* d_ws, size_t ws_size,
                              hipStream_t stream) {
  const float* x       = (const float*)d_in[0];
  const float* g_w     = (const float*)d_in[1];
  const float* g_b     = (const float*)d_in[2];
  const float* theta_w = (const float*)d_in[3];
  const float* theta_b = (const float*)d_in[4];
  const float* phi_w   = (const float*)d_in[5];
  const float* phi_b   = (const float*)d_in[6];
  const float* w_w     = (const float*)d_in[7];
  const float* w_b     = (const float*)d_in[8];
  const float* bn_g    = (const float*)d_in[9];
  const float* bn_b    = (const float*)d_in[10];
  const float* bn_m    = (const float*)d_in[11];
  const float* bn_v    = (const float*)d_in[12];
  float* out = (float*)d_out;

  char* ws = (char*)d_ws;
  size_t off = 0;
  auto alloc = [&](size_t bytes) -> void* {
    void* p = ws + off;
    off += (bytes + 255) & ~(size_t)255;
    return p;
  };

  bf16_t* xbf   = (bf16_t*)alloc((size_t)NP * NC * sizeof(bf16_t));    // 64 MB
  bf16_t* gwbf  = (bf16_t*)alloc((size_t)NIC * NC * sizeof(bf16_t));   // 1 MB
  bf16_t* twbf  = (bf16_t*)alloc((size_t)NIC * NC * sizeof(bf16_t));
  bf16_t* pwbf  = (bf16_t*)alloc((size_t)NIC * NC * sizeof(bf16_t));
  bf16_t* wwbf  = (bf16_t*)alloc((size_t)NC * NIC * sizeof(bf16_t));
  bf16_t* xpbf  = (bf16_t*)alloc((size_t)NBT * NC * sizeof(bf16_t));
  float*  thf   = (float*)alloc((size_t)NBT * NIC * sizeof(float));
  float*  phf   = (float*)alloc((size_t)NBT * NIC * sizeof(float));
  float*  attn  = (float*)alloc((size_t)NB * NT * NT * sizeof(float));
  bf16_t* gbuf  = (bf16_t*)alloc((size_t)NP * NIC * sizeof(bf16_t));   // 32 MB
  bf16_t* ybuf  = (bf16_t*)alloc((size_t)NP * NIC * sizeof(bf16_t));   // 32 MB
  (void)ws_size; (void)in_sizes; (void)n_in; (void)out_size;

  // weights -> bf16
  k_cvt_bf16<<<(NIC * NC + 255) / 256, 256, 0, stream>>>(g_w,     gwbf, NIC * NC);
  k_cvt_bf16<<<(NIC * NC + 255) / 256, 256, 0, stream>>>(theta_w, twbf, NIC * NC);
  k_cvt_bf16<<<(NIC * NC + 255) / 256, 256, 0, stream>>>(phi_w,   pwbf, NIC * NC);
  k_cvt_bf16<<<(NC * NIC + 255) / 256, 256, 0, stream>>>(w_w,     wwbf, NC * NIC);

  // transpose + pool
  k_transpose_pool<<<dim3(NC / 32, NBT), 256, 0, stream>>>(x, xbf, xpbf);

  // theta / phi GEMMs (bf16 WMMA)
  k_qk_gemm<<<dim3(NIC / 16, NBT / 16, 2), 32, 0, stream>>>(
      xpbf, twbf, pwbf, theta_b, phi_b, thf, phf);

  // attention scores + softmax
  k_attn<<<NB, 64, 0, stream>>>(thf, phf, attn);

  // g projection (bf16 WMMA, 2x4 register blocking, ic-major D)
  k_g_gemm<<<dim3(NP / 64, NIC / 32 / 8), 256, 0, stream>>>(xbf, gwbf, g_b, gbuf);

  // temporal attention mix
  k_mix<<<dim3((NHW * (NIC / 4)) / 256, NB), 256, 0, stream>>>(gbuf, attn, ybuf);

  // output projection + BN + residual (bf16 WMMA, 4x2 register blocking)
  k_out_gemm<<<dim3(NC / 32, NP / 64 / 8), 256, 0, stream>>>(
      ybuf, wwbf, w_b, bn_g, bn_b, bn_m, bn_v, x, out);
}
